// DiffuserJointer_72335839200103
// MI455X (gfx1250) — matrix operand order repaired and checked by reading, not compiled, not run on hardware
//
#include <hip/hip_runtime.h>
#include <math.h>

typedef __attribute__((ext_vector_type(16))) __bf16 v16bf;
typedef __attribute__((ext_vector_type(8)))  __bf16 v8bf;
typedef __attribute__((ext_vector_type(8)))  float  v8f;

#define Dm 384
#define Hh 8
#define Bb 8
#define Ls 64
#define Nc 2560
#define Kf 512
#define HD 48
#define SEQ (Ls + Kf)      /* 576  */
#define MT  (Ls * Bb)      /* 512  */
#define MC  (Nc * Bb)      /* 20480*/
#define MS  (SEQ * Bb)     /* 4608 */
#define LOG1E4 9.210340371976184f

// ---------------------------------------------------------------- conversions
__global__ void k_cvt_bf16(const float* __restrict__ s, __bf16* __restrict__ d, int n) {
  int i = blockIdx.x * blockDim.x + threadIdx.x;
  if (i < n) d[i] = (__bf16)s[i];
}

__global__ void k_copy(const float* __restrict__ s, float* __restrict__ d, int n) {
  int i = blockIdx.x * blockDim.x + threadIdx.x;
  if (i < n) d[i] = s[i];
}

// ---------------------------------------------------------------- WMMA GEMM
// C[M,384] = A(bf16)[M,384] x W(bf16)[384,384]^T + bias   (optional ReLU)
// N = K = 384 are compile-time constants -> K-loop (12 iters) fully unrolls,
// address math folds into immediate offsets, waits can be staged.
// wave32: each wave owns a 16(M) x 64(N) strip -> 4 accumulators, one A
// fragment feeds 4 WMMAs per K-step. 8 waves stacked in M -> 128x64 per block.
// Requires M%128==0 (all WMMA GEMMs here: 512 / 4608 / 20480).
__global__ __launch_bounds__(256) void k_wmma_gemm(
    const __bf16* __restrict__ A, const __bf16* __restrict__ W,
    const float* __restrict__ bias, float* __restrict__ C,
    int M, int relu)
{
  const int lane = threadIdx.x & 31;
  const int wave = threadIdx.x >> 5;
  const int m0 = blockIdx.y * 128 + wave * 16;
  const int n0 = blockIdx.x * 64;
  const int r  = lane & 15;            // A: row in tile; B: column in tile
  const int kh = lane >> 4;            // K-half select per ISA layout

  const __bf16* arow = A + (size_t)(m0 + r) * Dm + kh * 8;
  const __bf16* wrow = W + (size_t)(n0 + r) * Dm + kh * 8;   // column n0+r of B-tile 0

  v8f acc[4] = {{}, {}, {}, {}};
#pragma unroll
  for (int k0 = 0; k0 < Dm; k0 += 32) {
    // A fragment: lane holds k = {kh*8..+7} and {16+kh*8..+7} (16-bit A layout)
    const v8bf* ap = reinterpret_cast<const v8bf*>(arow + k0);
    v8bf alo = ap[0];
    v8bf ahi = ap[2];                  // +16 elements
    v16bf a;
#pragma unroll
    for (int e = 0; e < 8; ++e) { a[e] = alo[e]; a[8 + e] = ahi[e]; }
    if (k0 + 32 < Dm) {                // hint next K-step toward the caches
      __builtin_prefetch(arow + k0 + 32, 0, 3);
      __builtin_prefetch(wrow + k0 + 32, 0, 3);
    }
    // preload all four B fragments, then run the 4-WMMA chain (staged waits)
    v16bf bfr[4];
#pragma unroll
    for (int t = 0; t < 4; ++t) {
      // B fragment: column (n0 + t*16 + r), k = {kh*8..+7} and {16+kh*8..+7} in W row
      const v8bf* wp = reinterpret_cast<const v8bf*>(wrow + (size_t)t * 16 * Dm + k0);
      v8bf lo = wp[0], hi = wp[2];
#pragma unroll
      for (int e = 0; e < 8; ++e) { bfr[t][e] = lo[e]; bfr[t][8 + e] = hi[e]; }
    }
#pragma unroll
    for (int t = 0; t < 4; ++t)
      acc[t] = __builtin_amdgcn_wmma_f32_16x16x32_bf16(false, a, false, bfr[t], (short)0,
                                                       acc[t], false, false);
  }
  // C/D layout: VGPR i -> row kh*8 + i, column = lane&15
  const int row0 = m0 + kh * 8;
#pragma unroll
  for (int t = 0; t < 4; ++t) {
    const int col = n0 + t * 16 + r;
    const float bv = bias[col];
#pragma unroll
    for (int i = 0; i < 8; ++i) {
      float v = acc[t][i] + bv;
      if (relu) v = fmaxf(v, 0.f);
      C[(size_t)(row0 + i) * Dm + col] = v;
    }
  }
}

// ---------------------------------------------------------------- small GEMM
__global__ void k_naive(const float* __restrict__ A, const float* __restrict__ W,
                        const float* __restrict__ bias, float* __restrict__ C,
                        int M, int N, int Kd, int relu)
{
  int i = blockIdx.x * blockDim.x + threadIdx.x;
  if (i >= M * N) return;
  int m = i / N, n = i % N;
  const float* a = A + (size_t)m * Kd;
  const float* w = W + (size_t)n * Kd;
  float s = bias[n];
  for (int k = 0; k < Kd; ++k) s += a[k] * w[k];
  if (relu) s = fmaxf(s, 0.f);
  C[i] = s;
}

// ---------------------------------------------------------------- embeddings
__device__ inline float sinemb(float x, int d) {
  const int half = Dm / 2;
  int j = (d < half) ? d : (d - half);
  float f = __expf(-LOG1E4 / (float)(half - 1) * (float)j);
  float a = x * f;
  return (d < half) ? __sinf(a) : __cosf(a);
}

__global__ void k_embed_traj(const float* __restrict__ trajectory,
                             const float* __restrict__ tw, const float* __restrict__ tb,
                             float* __restrict__ out)
{
  int i = blockIdx.x * blockDim.x + threadIdx.x;
  if (i >= MT * Dm) return;
  int t = i / Dm, d = i % Dm;
  int s = t / Bb, b = t % Bb;
  const float* x = trajectory + ((size_t)b * Ls + s) * 7;
  float acc = tb[d];
#pragma unroll
  for (int k = 0; k < 7; ++k) acc += x[k] * tw[d * 7 + k];
  out[i] = acc + sinemb((float)s, d);
}

__global__ void k_sin_timestep(const int* __restrict__ ts, float* __restrict__ out) {
  int i = blockIdx.x * blockDim.x + threadIdx.x;
  if (i >= Bb * Dm) return;
  int b = i / Dm, d = i % Dm;
  out[i] = sinemb((float)ts[b], d);
}

// pe_all (B, SEQ, Dm, 2): first Ls from rope3d(trajectory xyz), rest from fps_pos
__global__ void k_build_pe(const float* __restrict__ trajectory,
                           const float* __restrict__ fps_pos, float* __restrict__ pe)
{
  int i = blockIdx.x * blockDim.x + threadIdx.x;
  if (i >= Bb * SEQ * Dm) return;
  int d = i % Dm;
  int rest = i / Dm;
  int s = rest % SEQ;
  int b = rest / SEQ;
  float c, sn;
  if (s < Ls) {
    int axis = d / 128, m = (d % 128) / 2;
    float x  = trajectory[((size_t)b * Ls + s) * 7 + axis];
    float dv = __expf(-LOG1E4 * (2.f * (float)m) / 128.f);
    float a  = x * dv;
    c = __cosf(a); sn = __sinf(a);
  } else {
    size_t o = (((size_t)b * Kf + (s - Ls)) * Dm + d) * 2;
    c = fps_pos[o]; sn = fps_pos[o + 1];
  }
  size_t po = (((size_t)b * SEQ + s) * Dm + d) * 2;
  pe[po] = c; pe[po + 1] = sn;
}

// rotate x (token rows s*B+b) with materialized pe (stride SEQ)
__global__ void k_rope_pe(float* __restrict__ x, const float* __restrict__ pe, int n) {
  int i = blockIdx.x * blockDim.x + threadIdx.x;   // pair index
  if (i >= n) return;
  int t = i / (Dm / 2), p = i % (Dm / 2);
  int s = t / Bb, b = t % Bb, d = 2 * p;
  size_t po = (((size_t)b * SEQ + s) * Dm + d) * 2;
  float c = pe[po], sn = pe[po + 1];
  size_t xo = (size_t)t * Dm + d;
  float x0 = x[xo], x1 = x[xo + 1];
  x[xo]     = x0 * c - x1 * sn;
  x[xo + 1] = x1 * c + x0 * sn;
}

// rotate context k (token rows b*Nc+j) with rope computed from xyz on the fly
__global__ void k_rope_ctx(float* __restrict__ x, const float* __restrict__ xyz, int n) {
  int i = blockIdx.x * blockDim.x + threadIdx.x;
  if (i >= n) return;
  int t = i / (Dm / 2), p = i % (Dm / 2);
  int axis = (2 * p) / 128, m = p % 64;
  float v  = xyz[(size_t)t * 3 + axis];
  float dv = __expf(-LOG1E4 * (2.f * (float)m) / 128.f);
  float a  = v * dv;
  float c = __cosf(a), sn = __sinf(a);
  size_t xo = (size_t)t * Dm + 2 * p;
  float x0 = x[xo], x1 = x[xo + 1];
  x[xo]     = x0 * c - x1 * sn;
  x[xo + 1] = x1 * c + x0 * sn;
}

// ---------------------------------------------------------------- AdaLN apply
__global__ void k_adaln(const float* __restrict__ x, const float* __restrict__ mod,
                        float* __restrict__ out, int n)
{
  int i = blockIdx.x * blockDim.x + threadIdx.x;
  if (i >= n) return;
  int t = i / Dm, d = i % Dm, b = t % Bb;
  float sc = mod[b * (2 * Dm) + d];
  float sh = mod[b * (2 * Dm) + Dm + d];
  out[i] = x[i] * (1.f + sc) + sh;
}

__global__ void k_add_silu(const float* __restrict__ a, const float* __restrict__ b,
                           float* __restrict__ o, int n)
{
  int i = blockIdx.x * blockDim.x + threadIdx.x;
  if (i >= n) return;
  float v = a[i] + b[i];
  o[i] = v / (1.f + __expf(-v));
}

// ---------------------------------------------------------------- attention
// one block per (qs, b, h); scores staged in LDS; two-phase softmax
__global__ __launch_bounds__(256) void k_attn(
    const float* __restrict__ q, const float* __restrict__ k, const float* __restrict__ v,
    float* __restrict__ out, int Lq, int Lk, int bnMajor)
{
  __shared__ float sc[2560];
  __shared__ float red[256];
  __shared__ float qv[HD];
  __shared__ float oacc[4][64];
  int blk = blockIdx.x;
  int h = blk % Hh;
  int rest = blk / Hh;
  int b = rest % Bb;
  int qs = rest / Bb;
  int tid = threadIdx.x;
  size_t qoff = ((size_t)qs * Bb + b) * Dm + h * HD;
  if (tid < HD) qv[tid] = q[qoff + tid];
  __syncthreads();

  const float scale = rsqrtf((float)HD);
  float lmax = -3.0e38f;
  for (int j = tid; j < Lk; j += 256) {
    size_t ko = (bnMajor ? ((size_t)b * Lk + j) : ((size_t)j * Bb + b)) * Dm + h * HD;
    const float4* kp = reinterpret_cast<const float4*>(k + ko);
    float s = 0.f;
#pragma unroll
    for (int d4 = 0; d4 < HD / 4; ++d4) {
      float4 kv4 = kp[d4];
      s += qv[4 * d4 + 0] * kv4.x + qv[4 * d4 + 1] * kv4.y +
           qv[4 * d4 + 2] * kv4.z + qv[4 * d4 + 3] * kv4.w;
    }
    s *= scale;
    sc[j] = s;
    lmax = fmaxf(lmax, s);
  }
  red[tid] = lmax; __syncthreads();
  for (int st = 128; st > 0; st >>= 1) {
    if (tid < st) red[tid] = fmaxf(red[tid], red[tid + st]);
    __syncthreads();
  }
  float m = red[0]; __syncthreads();

  float lsum = 0.f;
  for (int j = tid; j < Lk; j += 256) {
    float e = __expf(sc[j] - m);
    sc[j] = e;
    lsum += e;
  }
  red[tid] = lsum; __syncthreads();
  for (int st = 128; st > 0; st >>= 1) {
    if (tid < st) red[tid] += red[tid + st];
    __syncthreads();
  }
  float inv = 1.f / red[0];

  int d = tid & 63, g = tid >> 6;     // 4 j-groups x 64 dims (48 active)
  float acc = 0.f;
  if (d < HD) {
    for (int j = g; j < Lk; j += 4) {
      size_t vo = (bnMajor ? ((size_t)b * Lk + j) : ((size_t)j * Bb + b)) * Dm + h * HD;
      acc += sc[j] * v[vo + d];
    }
  }
  oacc[g][d] = acc; __syncthreads();
  if (g == 0 && d < HD)
    out[qoff + d] = (oacc[0][d] + oacc[1][d] + oacc[2][d] + oacc[3][d]) * inv;
}

// ---------------------------------------------------------------- residual+LN
// wave32 per row, 8 rows per block; in-place on x: x = LN(x + y)*g + be
__global__ __launch_bounds__(256) void k_add_ln(
    float* __restrict__ x, const float* __restrict__ y,
    const float* __restrict__ g, const float* __restrict__ be, int M)
{
  int wave = threadIdx.x >> 5, lane = threadIdx.x & 31;
  int row = blockIdx.x * 8 + wave;
  if (row >= M) return;
  const float* xr = x + (size_t)row * Dm;
  const float* yr = y + (size_t)row * Dm;
  float vals[12];
  float s = 0.f;
#pragma unroll
  for (int i = 0; i < 12; ++i) {
    int d = lane + 32 * i;
    vals[i] = xr[d] + yr[d];
    s += vals[i];
  }
  for (int off = 16; off > 0; off >>= 1) s += __shfl_xor(s, off, 32);
  float mean = s / (float)Dm;
  float vs = 0.f;
#pragma unroll
  for (int i = 0; i < 12; ++i) { float dd = vals[i] - mean; vs += dd * dd; }
  for (int off = 16; off > 0; off >>= 1) vs += __shfl_xor(vs, off, 32);
  float rstd = rsqrtf(vs / (float)Dm + 1e-5f);
  float* xw = x + (size_t)row * Dm;
#pragma unroll
  for (int i = 0; i < 12; ++i) {
    int d = lane + 32 * i;
    xw[d] = (vals[i] - mean) * rstd * g[d] + be[d];
  }
}

// ---------------------------------------------------------------- output heads
__global__ void k_heads(const float* __restrict__ hpp, const float* __restrict__ hop,
                        const float* __restrict__ pp2w, const float* __restrict__ pp2b,
                        const float* __restrict__ op2w, const float* __restrict__ op2b,
                        float* __restrict__ out)
{
  int i = blockIdx.x * blockDim.x + threadIdx.x;
  if (i >= Bb * Ls * 8) return;
  int ch = i % 8;
  int rest = i / 8;
  int l = rest % Ls;
  int b = rest / Ls;
  int t = l * Bb + b;
  float s;
  if (ch < 7) {
    s = pp2b[ch];
    const float* h = hpp + (size_t)t * Dm;
    const float* w = pp2w + (size_t)ch * Dm;
    for (int k = 0; k < Dm; ++k) s += h[k] * w[k];
  } else {
    s = op2b[0];
    const float* h = hop + (size_t)t * Dm;
    for (int k = 0; k < Dm; ++k) s += h[k] * op2w[k];
  }
  out[i] = s;
}

// ================================================================ host
extern "C" void kernel_launch(void* const* d_in, const int* in_sizes, int n_in,
                              void* d_out, int out_size, void* d_ws, size_t ws_size,
                              hipStream_t stream)
{
  (void)in_sizes; (void)n_in; (void)out_size; (void)ws_size;
  const float* trajectory  = (const float*)d_in[0];
  const int*   timestep    = (const int*)d_in[1];
  const float* context_fe  = (const float*)d_in[2];   // (B,N,D) -> token row b*Nc+j
  const float* context_xyz = (const float*)d_in[3];   // (B,N,3)
  const float* gripper     = (const float*)d_in[4];   // (B,NHIST*D) flat
  const float* fps_feats   = (const float*)d_in[5];   // (K,B,D) -> token row kk*Bb+b
  const float* fps_pos     = (const float*)d_in[6];   // (B,K,D,2)
  auto Pp = [&](int i) -> const float* { return (const float*)d_in[7 + i]; };

  float* fp = (float*)d_ws;
  size_t off = 0;
  auto alloc = [&](size_t n) -> float* { float* p = fp + off; off += n; return p; };
  float* traj   = alloc((size_t)MT * Dm);
  float* feats  = alloc((size_t)MS * Dm);
  float* posf   = alloc((size_t)MS * Dm);
  float* qb     = alloc((size_t)MS * Dm);
  float* kb     = alloc((size_t)MC * Dm);
  float* vb     = alloc((size_t)MC * Dm);
  float* attn_o = alloc((size_t)MS * Dm);
  float* tmp1   = alloc((size_t)MS * Dm);
  float* hbuf   = alloc((size_t)MS * Dm);
  float* pe_all = alloc((size_t)Bb * SEQ * Dm * 2);
  float* semb   = alloc(Bb * Dm);
  float* tA     = alloc(Bb * Dm);
  float* tfe    = alloc(Bb * Dm);
  float* gtmp   = alloc(Bb * Dm);
  float* gfe    = alloc(Bb * Dm);
  float* stemb  = alloc(Bb * Dm);
  float* modb   = alloc(Bb * 2 * Dm);
  float* pf     = alloc((size_t)MT * Dm);
  float* hpp    = alloc((size_t)MT * Dm);
  float* hop    = alloc((size_t)MT * Dm);
  off = (off + 7) & ~(size_t)7;                // 32B-align bf16 region
  __bf16* bfbase = (__bf16*)(fp + off);
  size_t bo = 0;
  auto ballo = [&](size_t n) -> __bf16* { __bf16* p = bfbase + bo; bo += n; return p; };
  __bf16* ctx_bf = ballo((size_t)MC * Dm);     // context features, converted once
  __bf16* xbf    = ballo((size_t)MS * Dm);     // q-side activation staging
  __bf16* kvbf   = ballo((size_t)MS * Dm);     // kv / hidden activation staging
  __bf16* wbf    = bfbase + bo;                // weights

  auto cvt = [&](const float* s, __bf16* d, int n) {
    k_cvt_bf16<<<dim3((n + 255) / 256), 256, 0, stream>>>(s, d, n);
  };

  // params: 0..9 pre, then 8 layers x 16 leaves, then 138.. heads
  struct LP { const float* p[16]; const __bf16 *in_bf, *out_bf, *w1_bf, *w2_bf; };
  LP lp[8];
  size_t wo = 0;
  for (int li = 0; li < 8; ++li) {
    int base = 10 + 16 * li;
    for (int j = 0; j < 16; ++j) lp[li].p[j] = Pp(base + j);
    lp[li].in_bf  = wbf + wo; cvt(lp[li].p[2],  wbf + wo, 3 * Dm * Dm); wo += (size_t)3 * Dm * Dm;
    lp[li].out_bf = wbf + wo; cvt(lp[li].p[4],  wbf + wo, Dm * Dm);     wo += (size_t)Dm * Dm;
    lp[li].w1_bf  = wbf + wo; cvt(lp[li].p[10], wbf + wo, Dm * Dm);     wo += (size_t)Dm * Dm;
    lp[li].w2_bf  = wbf + wo; cvt(lp[li].p[12], wbf + wo, Dm * Dm);     wo += (size_t)Dm * Dm;
  }
  __bf16* pproj_bf = wbf + wo; cvt(Pp(138), pproj_bf, Dm * Dm); wo += (size_t)Dm * Dm;
  __bf16* pp1_bf   = wbf + wo; cvt(Pp(140), pp1_bf,   Dm * Dm); wo += (size_t)Dm * Dm;
  __bf16* op1_bf   = wbf + wo; cvt(Pp(144), op1_bf,   Dm * Dm); wo += (size_t)Dm * Dm;

  auto gemm = [&](const __bf16* A, const __bf16* W, const float* bias, float* C, int M, int relu) {
    k_wmma_gemm<<<dim3(Dm / 64, M / 128), 256, 0, stream>>>(A, W, bias, C, M, relu);
  };
  auto ew = [&](int n) { return dim3((unsigned)((n + 255) / 256)); };

  // ---- timestep / gripper conditioning -> silu(t_emb)
  k_embed_traj<<<ew(MT * Dm), 256, 0, stream>>>(trajectory, Pp(0), Pp(1), traj);
  k_build_pe<<<ew(Bb * SEQ * Dm), 256, 0, stream>>>(trajectory, fps_pos, pe_all);
  k_sin_timestep<<<ew(Bb * Dm), 256, 0, stream>>>(timestep, semb);
  k_naive<<<ew(Bb * Dm), 256, 0, stream>>>(semb, Pp(2), Pp(3), tA, Bb, Dm, Dm, 1);
  k_naive<<<ew(Bb * Dm), 256, 0, stream>>>(tA, Pp(4), Pp(5), tfe, Bb, Dm, Dm, 0);
  k_naive<<<ew(Bb * Dm), 256, 0, stream>>>(gripper, Pp(6), Pp(7), gtmp, Bb, Dm, 3 * Dm, 1);
  k_naive<<<ew(Bb * Dm), 256, 0, stream>>>(gtmp, Pp(8), Pp(9), gfe, Bb, Dm, Dm, 0);
  k_add_silu<<<ew(Bb * Dm), 256, 0, stream>>>(tfe, gfe, stemb, Bb * Dm);
  cvt(context_fe, ctx_bf, MC * Dm);            // context kv source, converted once

  // ---- one attention block (AdaLN -> MHA -> LN -> AdaLN -> FFN -> LN)
  auto attn_block = [&](float* x, const float* kvsrc, const __bf16* kvsrc_bf,
                        int Mq, int Mk, int Lq, int Lk, int bnMajor, const LP& L) {
    const __bf16* kv_bf = kvsrc_bf;
    if (!kv_bf) { cvt(kvsrc, kvbf, Mk * Dm); kv_bf = kvbf; }  // raw feats -> bf16
    k_naive<<<ew(Bb * 2 * Dm), 256, 0, stream>>>(stemb, L.p[0], L.p[1], modb, Bb, 2 * Dm, Dm, 0);
    k_adaln<<<ew(Mq * Dm), 256, 0, stream>>>(x, modb, tmp1, Mq * Dm);
    cvt(tmp1, xbf, Mq * Dm);
    gemm(xbf, L.in_bf, L.p[3], qb, Mq, 0);
    gemm(kv_bf, L.in_bf + (size_t)Dm * Dm,     L.p[3] + Dm,     kb, Mk, 0);
    gemm(kv_bf, L.in_bf + (size_t)2 * Dm * Dm, L.p[3] + 2 * Dm, vb, Mk, 0);
    k_rope_pe<<<ew(Mq * (Dm / 2)), 256, 0, stream>>>(qb, pe_all, Mq * (Dm / 2));
    if (bnMajor)
      k_rope_ctx<<<ew(Mk * (Dm / 2)), 256, 0, stream>>>(kb, context_xyz, Mk * (Dm / 2));
    else
      k_rope_pe<<<ew(Mk * (Dm / 2)), 256, 0, stream>>>(kb, pe_all, Mk * (Dm / 2));
    k_attn<<<dim3((unsigned)(Lq * Bb * Hh)), 256, 0, stream>>>(qb, kb, vb, attn_o, Lq, Lk, bnMajor);
    cvt(attn_o, xbf, Mq * Dm);
    gemm(xbf, L.out_bf, L.p[5], tmp1, Mq, 0);
    k_add_ln<<<dim3((unsigned)((Mq + 7) / 8)), 256, 0, stream>>>(x, tmp1, L.p[6], L.p[7], Mq);
    k_naive<<<ew(Bb * 2 * Dm), 256, 0, stream>>>(stemb, L.p[8], L.p[9], modb, Bb, 2 * Dm, Dm, 0);
    k_adaln<<<ew(Mq * Dm), 256, 0, stream>>>(x, modb, tmp1, Mq * Dm);
    cvt(tmp1, xbf, Mq * Dm);
    gemm(xbf, L.w1_bf, L.p[11], hbuf, Mq, 1);
    cvt(hbuf, kvbf, Mq * Dm);                  // kv staging free again here
    gemm(kvbf, L.w2_bf, L.p[13], tmp1, Mq, 0);
    k_add_ln<<<dim3((unsigned)((Mq + 7) / 8)), 256, 0, stream>>>(x, tmp1, L.p[14], L.p[15], Mq);
  };

  // ---- 2 cross-attention layers (traj queries -> context kv)
  attn_block(traj, nullptr, ctx_bf, MT, MC, Ls, Nc, 1, lp[0]);
  attn_block(traj, nullptr, ctx_bf, MT, MC, Ls, Nc, 1, lp[1]);

  // ---- concat [traj; fps_feats] (token orders line up exactly)
  k_copy<<<ew(MT * Dm), 256, 0, stream>>>(traj, feats, MT * Dm);
  k_copy<<<ew(Kf * Bb * Dm), 256, 0, stream>>>(fps_feats, feats + (size_t)MT * Dm, Kf * Bb * Dm);

  // ---- 4 self-attention layers
  for (int li = 2; li <= 5; ++li)
    attn_block(feats, feats, nullptr, MS, MS, SEQ, SEQ, 0, lp[li]);

  // ---- position head: 2 more self-attn layers on a copy
  k_copy<<<ew(MS * Dm), 256, 0, stream>>>(feats, posf, MS * Dm);
  for (int li = 6; li <= 7; ++li)
    attn_block(posf, posf, nullptr, MS, MS, SEQ, SEQ, 0, lp[li]);

  // ---- prediction MLPs (first Ls*Bb token rows == pos_f[:L])
  cvt(posf, xbf, MT * Dm);
  gemm(xbf, pproj_bf, Pp(139), pf, MT, 0);
  cvt(pf, xbf, MT * Dm);
  gemm(xbf, pp1_bf, Pp(141), hpp, MT, 1);
  gemm(xbf, op1_bf, Pp(145), hop, MT, 1);
  k_heads<<<ew(Bb * Ls * 8), 256, 0, stream>>>(hpp, hop, Pp(142), Pp(143), Pp(146), Pp(147),
                                               (float*)d_out);
}
